// AttentionXL_28621662060622
// MI455X (gfx1250) — matrix operand order repaired and checked by reading, not compiled, not run on hardware
//
#include <hip/hip_runtime.h>

#define CUR    1024
#define FULLK  2048
#define BSZ    4
#define DMODEL 1024
#define HN     16
#define HD     64
#define PREV   (FULLK - CUR)
#define NH     (HN * HD)      // 1024
#define ATT_SCALE 0.125f      // 1/sqrt(64)

typedef __attribute__((ext_vector_type(16))) __bf16         v16bf;
typedef __attribute__((ext_vector_type(8)))  float          v8f;
typedef __attribute__((ext_vector_type(8)))  unsigned short us8;

static __device__ __forceinline__ unsigned short f2bf(float f) {
  unsigned int u = __float_as_uint(f);
  unsigned int r = u + 0x7FFFu + ((u >> 16) & 1u);   // round-to-nearest-even
  return (unsigned short)(r >> 16);
}

static __device__ __forceinline__ v8f wmma_bf16(v16bf a, v16bf b, v8f c) {
  return __builtin_amdgcn_wmma_f32_16x16x32_bf16(
      /*neg_a=*/false, a, /*neg_b=*/false, b,
      /*c_mod=*/(short)0, c, /*reuse_a=*/false, /*reuse_b=*/false);
}

// A fragment (16x32, M x K) from row-major bf16-bits matrix with leading dim lda.
// ISA layout: lane = row (l&15); lanes 0-15 hold K 0..7 / 16..23, lanes 16-31 hold K 8..15 / 24..31.
static __device__ __forceinline__ v16bf load_a(const unsigned short* A, int lda,
                                               int row0, int k0, int lane) {
  int row = row0 + (lane & 15);
  int kh  = ((lane >> 4) & 1) << 3;           // 0 or 8
  const unsigned short* p = A + (size_t)row * lda + (k0 + kh);
  union { us8 h[2]; v16bf v; } u;
  u.h[0] = *reinterpret_cast<const us8*>(p);        // K = kh .. kh+7
  u.h[1] = *reinterpret_cast<const us8*>(p + 16);   // K = 16+kh .. 16+kh+7
  return u.v;
}

// B fragment (32x16, K x N) where B(k,n) = Bt[n][k]; Bt row-major, leading dim ldk.
// ISA layout: lane = col (l&15); lanes 0-15 hold K 0..7 / 16..23, lanes 16-31 hold K 8..15 / 24..31.
static __device__ __forceinline__ v16bf load_bt(const unsigned short* Bt, int ldk,
                                                int n0, int k0, int lane) {
  int row = n0 + (lane & 15);
  int kh  = ((lane >> 4) & 1) << 3;           // 0 or 8
  const unsigned short* p = Bt + (size_t)row * ldk + (k0 + kh);
  union { us8 h[2]; v16bf v; } u;
  u.h[0] = *reinterpret_cast<const us8*>(p);
  u.h[1] = *reinterpret_cast<const us8*>(p + 16);
  return u.v;
}

// Row reductions across the 16 lanes of a wave32 half (xor masks stay inside the half).
static __device__ __forceinline__ float hmax16(float v) {
#pragma unroll
  for (int m = 1; m < 16; m <<= 1) v = fmaxf(v, __shfl_xor(v, m, 32));
  return v;
}
static __device__ __forceinline__ float hsum16(float v) {
#pragma unroll
  for (int m = 1; m < 16; m <<= 1) v += __shfl_xor(v, m, 32);
  return v;
}

// ---------------- conversion kernels ----------------
__global__ void cvt_kernel(const float* __restrict__ src,
                           unsigned short* __restrict__ dst, int n) {
  int i = blockIdx.x * blockDim.x + threadIdx.x;
  if (i < n) dst[i] = f2bf(src[i]);
}

// src: [K][N] f32 -> dst: [N][K] bf16 bits (transposed, so B loads are contiguous in K)
__global__ void cvt_t_kernel(const float* __restrict__ src,
                             unsigned short* __restrict__ dst, int K, int N) {
  int i = blockIdx.x * blockDim.x + threadIdx.x;
  if (i < K * N) {
    int n = i / K, k = i - n * K;
    dst[i] = f2bf(src[(size_t)k * N + n]);
  }
}

// ---------------- GEMM main loop: 32x32 register tile per wave ----------------
static __device__ __forceinline__ void gemm_tile_32x32(
    const unsigned short* A, int lda, const unsigned short* Bt, int ldb,
    int m0, int n0, int K, int lane, v8f acc[2][2]) {
  for (int k0 = 0; k0 < K; k0 += 32) {
    v16bf a0 = load_a(A, lda, m0,      k0, lane);
    v16bf a1 = load_a(A, lda, m0 + 16, k0, lane);
    v16bf b0 = load_bt(Bt, ldb, n0,      k0, lane);
    v16bf b1 = load_bt(Bt, ldb, n0 + 16, k0, lane);
    acc[0][0] = wmma_bf16(a0, b0, acc[0][0]);
    acc[0][1] = wmma_bf16(a0, b1, acc[0][1]);
    acc[1][0] = wmma_bf16(a1, b0, acc[1][0]);
    acc[1][1] = wmma_bf16(a1, b1, acc[1][1]);
  }
}

// KV projection: A=[FULL*BS][D] full_input(bf16), Bt=[2048][D] W_kv^T.
// Epilogue: +b_kv, split into K [b,h,full,hd] and V transposed [b,h,hd,full].
__global__ void __launch_bounds__(128) gemm_kv_kernel(
    const unsigned short* __restrict__ A, const unsigned short* __restrict__ Bt,
    const float* __restrict__ bias,
    unsigned short* __restrict__ Kbf, unsigned short* __restrict__ Vt) {
  int lane = threadIdx.x & 31, wave = threadIdx.x >> 5;
  int m0 = blockIdx.y * 32;
  int n0 = blockIdx.x * 128 + wave * 32;
  v8f acc[2][2] = {};
  gemm_tile_32x32(A, DMODEL, Bt, DMODEL, m0, n0, DMODEL, lane, acc);
  int half = (lane >> 4) & 1, lc = lane & 15;
#pragma unroll
  for (int tn = 0; tn < 2; tn++) {
    int n = n0 + tn * 16 + lc;
    float bn = bias[n];
    int nk = (n < NH) ? n : n - NH;
    int hh = nk >> 6, hd = nk & 63;
#pragma unroll
    for (int tm = 0; tm < 2; tm++) {
#pragma unroll
      for (int r = 0; r < 8; r++) {
        int row = m0 + tm * 16 + r + 8 * half;   // row = f*BS + b
        int f = row >> 2, bb = row & 3;
        unsigned short vbits = f2bf(acc[tm][tn][r] + bn);
        if (n < NH)
          Kbf[(((size_t)(bb * HN + hh)) * FULLK + f) * HD + hd] = vbits;
        else
          Vt[(((size_t)(bb * HN + hh)) * HD + hd) * FULLK + f] = vbits;
      }
    }
  }
}

// Q projection with fused bias + u/v add -> QU, QV as [b,h,cur,hd] bf16.
__global__ void __launch_bounds__(128) gemm_q_kernel(
    const unsigned short* __restrict__ A, const unsigned short* __restrict__ Bt,
    const float* __restrict__ bias, const float* __restrict__ u_,
    const float* __restrict__ v_,
    unsigned short* __restrict__ QU, unsigned short* __restrict__ QV) {
  int lane = threadIdx.x & 31, wave = threadIdx.x >> 5;
  int m0 = blockIdx.y * 32;
  int n0 = blockIdx.x * 128 + wave * 32;
  v8f acc[2][2] = {};
  gemm_tile_32x32(A, DMODEL, Bt, DMODEL, m0, n0, DMODEL, lane, acc);
  int half = (lane >> 4) & 1, lc = lane & 15;
#pragma unroll
  for (int tn = 0; tn < 2; tn++) {
    int n = n0 + tn * 16 + lc;
    float bn = bias[n], un = u_[n], vn = v_[n];
    int hh = n >> 6, hd = n & 63;
#pragma unroll
    for (int tm = 0; tm < 2; tm++) {
#pragma unroll
      for (int r = 0; r < 8; r++) {
        int row = m0 + tm * 16 + r + 8 * half;   // row = i*BS + b
        int i = row >> 2, bb = row & 3;
        float q = acc[tm][tn][r] + bn;
        size_t idx = (((size_t)(bb * HN + hh)) * CUR + i) * HD + hd;
        QU[idx] = f2bf(q + un);
        QV[idx] = f2bf(q + vn);
      }
    }
  }
}

// R projection -> [h][full][hd] bf16.
__global__ void __launch_bounds__(128) gemm_r_kernel(
    const unsigned short* __restrict__ A, const unsigned short* __restrict__ Bt,
    const float* __restrict__ bias, unsigned short* __restrict__ Rbf) {
  int lane = threadIdx.x & 31, wave = threadIdx.x >> 5;
  int m0 = blockIdx.y * 32;
  int n0 = blockIdx.x * 128 + wave * 32;
  v8f acc[2][2] = {};
  gemm_tile_32x32(A, DMODEL, Bt, DMODEL, m0, n0, DMODEL, lane, acc);
  int half = (lane >> 4) & 1, lc = lane & 15;
#pragma unroll
  for (int tn = 0; tn < 2; tn++) {
    int n = n0 + tn * 16 + lc;
    float bn = bias[n];
    int hh = n >> 6, hd = n & 63;
#pragma unroll
    for (int tm = 0; tm < 2; tm++) {
#pragma unroll
      for (int r = 0; r < 8; r++) {
        int f = m0 + tm * 16 + r + 8 * half;
        Rbf[(((size_t)hh) * FULLK + f) * HD + hd] = f2bf(acc[tm][tn][r] + bn);
      }
    }
  }
}

// Output projection: A=O[cur*bs][NH], Bt=[D][NH] W_proj^T -> f32 d_out (+b_proj).
__global__ void __launch_bounds__(128) gemm_out_kernel(
    const unsigned short* __restrict__ A, const unsigned short* __restrict__ Bt,
    const float* __restrict__ bias, float* __restrict__ out) {
  int lane = threadIdx.x & 31, wave = threadIdx.x >> 5;
  int m0 = blockIdx.y * 32;
  int n0 = blockIdx.x * 128 + wave * 32;
  v8f acc[2][2] = {};
  gemm_tile_32x32(A, NH, Bt, NH, m0, n0, NH, lane, acc);
  int half = (lane >> 4) & 1, lc = lane & 15;
#pragma unroll
  for (int tn = 0; tn < 2; tn++) {
    int n = n0 + tn * 16 + lc;
    float bn = bias[n];
#pragma unroll
    for (int tm = 0; tm < 2; tm++) {
#pragma unroll
      for (int r = 0; r < 8; r++) {
        int row = m0 + tm * 16 + r + 8 * half;
        out[(size_t)row * DMODEL + n] = acc[tm][tn][r] + bn;
      }
    }
  }
}

// ---------------- flash-style TXL attention: one wave per 16-row query strip ----
// rel_shift(position)[i,j] == (q_i+v)·r_{j-i+CUR-1} inside the causal mask.
__global__ void __launch_bounds__(32) attn_kernel(
    const unsigned short* __restrict__ QU, const unsigned short* __restrict__ QV,
    const unsigned short* __restrict__ Kbf, const unsigned short* __restrict__ Vt,
    const unsigned short* __restrict__ Rbf, unsigned short* __restrict__ Obf) {
  __shared__ float          Ptile[16 * 64];   // positional band for this j-step
  __shared__ unsigned short stage[16 * 32];   // exp-score re-layout C->A fragment

  const int lane = threadIdx.x;
  const int bid = blockIdx.x;
  const int it = bid & (CUR / 16 - 1);
  const int h  = (bid >> 6) & (HN - 1);
  const int b  = bid >> 10;
  const int i0 = it * 16;

  const unsigned short* QUs = QU + (((size_t)(b * HN + h)) * CUR + i0) * HD;
  const unsigned short* QVs = QV + (((size_t)(b * HN + h)) * CUR + i0) * HD;
  const unsigned short* Kb  = Kbf + ((size_t)(b * HN + h)) * FULLK * HD;
  const unsigned short* Vb  = Vt  + ((size_t)(b * HN + h)) * HD * FULLK;
  const unsigned short* Rb  = Rbf + ((size_t)h) * FULLK * HD;

  // Query fragments are loop-invariant: load once.
  const v16bf aqu0 = load_a(QUs, HD, 0, 0,  lane);
  const v16bf aqu1 = load_a(QUs, HD, 0, 32, lane);
  const v16bf aqv0 = load_a(QVs, HD, 0, 0,  lane);
  const v16bf aqv1 = load_a(QVs, HD, 0, 32, lane);

  v8f O0 = {}, O1 = {}, O2 = {}, O3 = {};
  float rmax[8], rsum[8];
#pragma unroll
  for (int r = 0; r < 8; r++) { rmax[r] = -1e30f; rsum[r] = 0.0f; }

  const int half = lane >> 4;
  const int lc = lane & 15;
  const int jend = i0 + 16 + PREV;            // last unmasked key + 1 (<= FULLK)

  for (int j0 = 0; j0 < jend; j0 += 32) {
    // Positional band P[m, d], d in [dt0, dt0+64): covers d = j-i+CUR-1 for this step.
    const int dt0 = j0 - i0 + (CUR - 16);     // >= 0, multiple of 16
#pragma unroll
    for (int dt = 0; dt < 4; dt++) {
      const int d0 = dt0 + dt * 16;
      v8f pacc = {};
      if (d0 < FULLK) {                       // wave-uniform branch
        pacc = wmma_bf16(aqv0, load_bt(Rb, HD, d0, 0,  lane), pacc);
        pacc = wmma_bf16(aqv1, load_bt(Rb, HD, d0, 32, lane), pacc);
      }
#pragma unroll
      for (int r = 0; r < 8; r++)
        Ptile[(r + 8 * half) * 64 + dt * 16 + lc] = pacc[r];
    }

    // Content scores, two 16-col tiles (32 keys).
    v8f c0 = {}, c1 = {};
    c0 = wmma_bf16(aqu0, load_bt(Kb, HD, j0,      0,  lane), c0);
    c0 = wmma_bf16(aqu1, load_bt(Kb, HD, j0,      32, lane), c0);
    c1 = wmma_bf16(aqu0, load_bt(Kb, HD, j0 + 16, 0,  lane), c1);
    c1 = wmma_bf16(aqu1, load_bt(Kb, HD, j0 + 16, 32, lane), c1);

    // Combine + shift-gather + mask + online softmax (per-row over lane halves).
#pragma unroll
    for (int r = 0; r < 8; r++) {
      const int m = r + 8 * half;
      const int i = i0 + m;
      const int ja = j0 + lc;
      const int jb = j0 + 16 + lc;
      float s0 = -1e30f, s1 = -1e30f;
      if (ja <= i + PREV) s0 = (c0[r] + Ptile[m * 64 + (ja - i + (CUR - 1) - dt0)]) * ATT_SCALE;
      if (jb <= i + PREV) s1 = (c1[r] + Ptile[m * 64 + (jb - i + (CUR - 1) - dt0)]) * ATT_SCALE;
      const float tmax = hmax16(fmaxf(s0, s1));
      const float nmax = fmaxf(rmax[r], tmax);
      const float fac  = __expf(rmax[r] - nmax);
      rmax[r] = nmax;
      const float e0 = __expf(s0 - nmax);
      const float e1 = __expf(s1 - nmax);
      rsum[r] = rsum[r] * fac + hsum16(e0 + e1);
      O0[r] *= fac; O1[r] *= fac; O2[r] *= fac; O3[r] *= fac;
      stage[m * 32 + lc]      = f2bf(e0);
      stage[m * 32 + 16 + lc] = f2bf(e1);
    }

    // Re-layout exp-scores (C-fragment) into a 16x32 A-fragment through LDS.
    union { unsigned short s[16]; v16bf v; } ua;
    {
      const int arow = lc;
      const int kh = half << 3;
#pragma unroll
      for (int e = 0; e < 8; e++) ua.s[e]     = stage[arow * 32 + kh + e];
#pragma unroll
      for (int e = 0; e < 8; e++) ua.s[8 + e] = stage[arow * 32 + 16 + kh + e];
    }

    // O += exp(S) @ V  (V stored transposed -> contiguous B loads).
    O0 = wmma_bf16(ua.v, load_bt(Vb, FULLK, 0,  j0, lane), O0);
    O1 = wmma_bf16(ua.v, load_bt(Vb, FULLK, 16, j0, lane), O1);
    O2 = wmma_bf16(ua.v, load_bt(Vb, FULLK, 32, j0, lane), O2);
    O3 = wmma_bf16(ua.v, load_bt(Vb, FULLK, 48, j0, lane), O3);
  }

  // Normalize and store O as bf16 rows (i, b, h*64+hd) for the final projection.
#pragma unroll
  for (int r = 0; r < 8; r++) {
    const int m = r + 8 * half;
    const float inv = 1.0f / rsum[r];
    unsigned short* dst = Obf + (((size_t)(i0 + m)) * BSZ + b) * NH + h * HD;
    dst[0  + lc] = f2bf(O0[r] * inv);
    dst[16 + lc] = f2bf(O1[r] * inv);
    dst[32 + lc] = f2bf(O2[r] * inv);
    dst[48 + lc] = f2bf(O3[r] * inv);
  }
}

// ---------------- host-side orchestration ----------------
extern "C" void kernel_launch(void* const* d_in, const int* in_sizes, int n_in,
                              void* d_out, int out_size, void* d_ws, size_t ws_size,
                              hipStream_t stream) {
  (void)in_sizes; (void)n_in; (void)out_size; (void)ws_size;
  const float* inputs  = (const float*)d_in[0];
  const float* pos_emb = (const float*)d_in[1];
  const float* full_in = (const float*)d_in[2];
  const float* u_      = (const float*)d_in[3];
  const float* v_      = (const float*)d_in[4];
  // d_in[5] = mask: causal structure computed analytically on device
  const float* W_kv   = (const float*)d_in[6];
  const float* b_kv   = (const float*)d_in[7];
  const float* W_q    = (const float*)d_in[8];
  const float* b_q    = (const float*)d_in[9];
  const float* W_pos  = (const float*)d_in[10];
  const float* b_pos  = (const float*)d_in[11];
  const float* W_proj = (const float*)d_in[12];
  const float* b_proj = (const float*)d_in[13];
  float* out = (float*)d_out;

  char* w = (char*)d_ws;
  auto alloc_u16 = [&](size_t elems) -> unsigned short* {
    unsigned short* p = (unsigned short*)w;
    w += elems * sizeof(unsigned short);
    return p;
  };
  unsigned short* in_bf   = alloc_u16((size_t)CUR * BSZ * DMODEL);
  unsigned short* full_bf = alloc_u16((size_t)FULLK * BSZ * DMODEL);
  unsigned short* pos_bf  = alloc_u16((size_t)FULLK * DMODEL);
  unsigned short* WkvT    = alloc_u16((size_t)2 * NH * DMODEL);
  unsigned short* WqT     = alloc_u16((size_t)NH * DMODEL);
  unsigned short* WposT   = alloc_u16((size_t)NH * DMODEL);
  unsigned short* WprojT  = alloc_u16((size_t)DMODEL * NH);
  unsigned short* Kbf     = alloc_u16((size_t)BSZ * HN * FULLK * HD);
  unsigned short* Vt      = alloc_u16((size_t)BSZ * HN * HD * FULLK);
  unsigned short* QU      = alloc_u16((size_t)BSZ * HN * CUR * HD);
  unsigned short* QV      = alloc_u16((size_t)BSZ * HN * CUR * HD);
  unsigned short* Rbf     = alloc_u16((size_t)HN * FULLK * HD);
  unsigned short* Obf     = alloc_u16((size_t)CUR * BSZ * NH);

  const int T = 256;
  cvt_kernel<<<(CUR * BSZ * DMODEL + T - 1) / T, T, 0, stream>>>(inputs, in_bf, CUR * BSZ * DMODEL);
  cvt_kernel<<<(FULLK * BSZ * DMODEL + T - 1) / T, T, 0, stream>>>(full_in, full_bf, FULLK * BSZ * DMODEL);
  cvt_kernel<<<(FULLK * DMODEL + T - 1) / T, T, 0, stream>>>(pos_emb, pos_bf, FULLK * DMODEL);
  cvt_t_kernel<<<(2 * NH * DMODEL + T - 1) / T, T, 0, stream>>>(W_kv, WkvT, DMODEL, 2 * NH);
  cvt_t_kernel<<<(NH * DMODEL + T - 1) / T, T, 0, stream>>>(W_q, WqT, DMODEL, NH);
  cvt_t_kernel<<<(NH * DMODEL + T - 1) / T, T, 0, stream>>>(W_pos, WposT, DMODEL, NH);
  cvt_t_kernel<<<(NH * DMODEL + T - 1) / T, T, 0, stream>>>(W_proj, WprojT, NH, DMODEL);

  gemm_kv_kernel<<<dim3((2 * NH) / 128, (FULLK * BSZ) / 32), 128, 0, stream>>>(
      full_bf, WkvT, b_kv, Kbf, Vt);
  gemm_q_kernel<<<dim3(NH / 128, (CUR * BSZ) / 32), 128, 0, stream>>>(
      in_bf, WqT, b_q, u_, v_, QU, QV);
  gemm_r_kernel<<<dim3(NH / 128, FULLK / 32), 128, 0, stream>>>(
      pos_bf, WposT, b_pos, Rbf);

  attn_kernel<<<BSZ * HN * (CUR / 16), 32, 0, stream>>>(QU, QV, Kbf, Vt, Rbf, Obf);

  gemm_out_kernel<<<dim3(DMODEL / 128, (CUR * BSZ) / 32), 128, 0, stream>>>(
      Obf, WprojT, b_proj, out);
}
